// NewEncoderBlock_15539191677175
// MI455X (gfx1250) — compile-verified, matrix operand order checked by reading, not run on hardware
//
#include <hip/hip_runtime.h>
#include <hip/hip_bf16.h>
#include <math.h>

// ---------------- problem constants (from reference setup_inputs) ----------
#define B_    4
#define CIN   128
#define COUT  256
#define H_    48
#define W_    48
#define L_    (H_*W_)     // 2304
#define DBR   32          // branch channels (Cin/4)
#define DI    64          // vss inner dim (2*DBR)
#define NST   16          // SSM state size
#define KDIR  4           // scan directions
#define RLOW  2           // dt low-rank

typedef __attribute__((ext_vector_type(16))) _Float16 v16h;
typedef __attribute__((ext_vector_type(8)))  _Float16 v8h;
typedef __attribute__((ext_vector_type(4)))  _Float16 v4h;
typedef __attribute__((ext_vector_type(8)))  float    v8f;
typedef __attribute__((ext_vector_type(4)))  float    v4f;

__device__ __forceinline__ float sigmoidf_(float x){ return 1.f/(1.f+__expf(-x)); }
__device__ __forceinline__ float siluf_(float x){ return x*sigmoidf_(x); }

// ======================================================================
// Batched GEMM via v_wmma_f32_16x16x32_f16.
//   C[bz] (M x N, ldc=N) = A[bz] (M x K, lda=K) * B[bz] (K x N, ldb=N)
//   transB: B stored (N x K), read B[n*K+k].  biasMode: 0 none, 1 per-N, 2 per-M
// 256 threads = 8 waves; block computes 64x64 C tile; each wave a 16x32
// strip (2 f32 accumulators, 2 WMMA per K-step, A-fragment reused).
// A and B staged to LDS as f16; B stored transposed (lBt[n][k]) so every
// fragment is two ds_load_b128 per the CDNA5 16-bit fragment layout.
// ======================================================================
#define TM 64
#define TN 64
#define TK 32
#define LDP (TK + 8)   // f16 row pitch: 40 f16 = 80B, 16B-aligned

__global__ __launch_bounds__(256)
void k_gemm_wmma(const float* __restrict__ A, const float* __restrict__ Bm,
                 float* __restrict__ C,
                 int M, int N, int K,
                 long long sA, long long sB, long long sC,
                 const float* __restrict__ bias, int biasMode, int transB)
{
  __shared__ _Float16 lA[TM][LDP];    // [m][k]
  __shared__ _Float16 lBt[TN][LDP];   // [n][k]  (B transposed in LDS)
  const int bz = blockIdx.z;
  const float* Ab = A  + (size_t)bz*sA;
  const float* Bb = Bm + (size_t)bz*sB;
  float*       Cb = C  + (size_t)bz*sC;
  const int m0 = blockIdx.y*TM;
  const int n0 = blockIdx.x*TN;
  const int tid  = threadIdx.x;
  const int lane = tid & 31;
  const int wave = tid >> 5;                 // 0..7
  const int wm = (wave >> 1) * 16;           // 0,16,32,48
  const int wn = (wave & 1) * 32;            // 0,32
  const int hl = lane >> 4, lr = lane & 15;  // lane half / lane row

  v8f acc0 = {}, acc1 = {};

  for (int k0 = 0; k0 < K; k0 += TK) {
    // prefetch next A K-tile (-> global_prefetch_b8)
    if (k0 + TK < K) __builtin_prefetch(Ab + (size_t)m0*K + k0 + TK, 0, 3);

    // ---- stage A: 64x32 f32 -> f16, vectorized float4 loads ----
    for (int t = tid; t < (TM*TK)/4; t += 256) {
      int r  = t >> 3;            // row in tile (TK/4 = 8 groups per row)
      int cg = (t & 7) * 4;       // col group
      int mm = m0 + r, kk = k0 + cg;
      v4f v = {0.f, 0.f, 0.f, 0.f};
      if (mm < M) {
        if (kk + 3 < K) {
          v = *(const v4f*)(Ab + (size_t)mm*K + kk);
        } else {
          #pragma unroll
          for (int j = 0; j < 4; ++j) if (kk + j < K) v[j] = Ab[(size_t)mm*K + kk + j];
        }
      }
      v4h hv;
      #pragma unroll
      for (int j = 0; j < 4; ++j) hv[j] = (_Float16)v[j];
      *(v4h*)&lA[r][cg] = hv;
    }
    // ---- stage B (transposed into lBt[n][k]) ----
    if (!transB) {
      // B is (K x N): vector-load along n, scatter to 4 LDS rows
      for (int t = tid; t < (TK*TN)/4; t += 256) {
        int r  = t >> 4;           // k row (TN/4 = 16 groups per row)
        int cg = (t & 15) * 4;     // n group
        int kk = k0 + r, nn = n0 + cg;
        v4f v = {0.f, 0.f, 0.f, 0.f};
        if (kk < K) {
          if (nn + 3 < N) {
            v = *(const v4f*)(Bb + (size_t)kk*N + nn);
          } else {
            #pragma unroll
            for (int j = 0; j < 4; ++j) if (nn + j < N) v[j] = Bb[(size_t)kk*N + nn + j];
          }
        }
        #pragma unroll
        for (int j = 0; j < 4; ++j) lBt[cg + j][r] = (_Float16)v[j];
      }
    } else {
      // B is (N x K): vector-load along k, contiguous LDS store
      for (int t = tid; t < (TN*TK)/4; t += 256) {
        int r  = t >> 3;           // n row (TK/4 = 8 groups per row)
        int cg = (t & 7) * 4;      // k group
        int nn = n0 + r, kk = k0 + cg;
        v4f v = {0.f, 0.f, 0.f, 0.f};
        if (nn < N) {
          if (kk + 3 < K) {
            v = *(const v4f*)(Bb + (size_t)nn*K + kk);
          } else {
            #pragma unroll
            for (int j = 0; j < 4; ++j) if (kk + j < K) v[j] = Bb[(size_t)nn*K + kk + j];
          }
        }
        v4h hv;
        #pragma unroll
        for (int j = 0; j < 4; ++j) hv[j] = (_Float16)v[j];
        *(v4h*)&lBt[r][cg] = hv;
      }
    }
    __syncthreads();

    // ---- fragments: two ds_load_b128 each (K contiguous per lane) ----
    // CDNA5 16-bit layout: e<8 -> K = 8*hl+e ; e>=8 -> K = 16+8*hl+(e-8)
    v8h aL = *(const v8h*)&lA[wm + lr][hl*8];
    v8h aH = *(const v8h*)&lA[wm + lr][16 + hl*8];
    v16h af = __builtin_shufflevector(aL, aH, 0,1,2,3,4,5,6,7,8,9,10,11,12,13,14,15);

    v8h b0L = *(const v8h*)&lBt[wn + lr][hl*8];
    v8h b0H = *(const v8h*)&lBt[wn + lr][16 + hl*8];
    v16h bf0 = __builtin_shufflevector(b0L, b0H, 0,1,2,3,4,5,6,7,8,9,10,11,12,13,14,15);

    v8h b1L = *(const v8h*)&lBt[wn + 16 + lr][hl*8];
    v8h b1H = *(const v8h*)&lBt[wn + 16 + lr][16 + hl*8];
    v16h bf1 = __builtin_shufflevector(b1L, b1H, 0,1,2,3,4,5,6,7,8,9,10,11,12,13,14,15);

    acc0 = __builtin_amdgcn_wmma_f32_16x16x32_f16(false, af, false, bf0,
                                                  (short)0, acc0, false, false);
    acc1 = __builtin_amdgcn_wmma_f32_16x16x32_f16(false, af, false, bf1,
                                                  (short)0, acc1, false, false);
    __syncthreads();
  }

  // ---- store: D VGPR r -> row (8*hl + r), col = lane&15 ----
  #pragma unroll
  for (int r = 0; r < 8; ++r) {
    int mm = m0 + wm + hl*8 + r;
    if (mm >= M) continue;
    int nnA = n0 + wn + lr;
    int nnB = n0 + wn + 16 + lr;
    if (nnA < N) {
      float v = acc0[r];
      if      (biasMode == 1) v += bias[nnA];
      else if (biasMode == 2) v += bias[mm];
      Cb[(size_t)mm*N + nnA] = v;
    }
    if (nnB < N) {
      float v = acc1[r];
      if      (biasMode == 1) v += bias[nnB];
      else if (biasMode == 2) v += bias[mm];
      Cb[(size_t)mm*N + nnB] = v;
    }
  }
  __builtin_amdgcn_s_wait_tensorcnt(0);     // CDNA5 split-counter wait path
#if __has_builtin(__builtin_amdgcn_s_wait_asynccnt)
  __builtin_amdgcn_s_wait_asynccnt(0);
#endif
}

// ---------------- channel slice: (B,CIN,L) -> contiguous (B,32,L) ----------
__global__ void k_slice(const float* __restrict__ x, float* __restrict__ o, int c0){
  int idx = blockIdx.x*blockDim.x + threadIdx.x;
  if (idx >= B_*DBR*L_) return;
  int pos = idx % L_; int c = (idx/L_)%DBR; int bb = idx/(DBR*L_);
  o[idx] = x[((size_t)bb*CIN + c0 + c)*L_ + pos];
}

// ---------------- layernorm over channels of NCHW -> row-major (B*L, C) ----
__global__ void k_ln_nchw_rows(const float* __restrict__ x, long long chanStride,
                               long long batchStride,
                               const float* __restrict__ g, const float* __restrict__ b,
                               float* __restrict__ out, int C)
{
  int idx = blockIdx.x*blockDim.x + threadIdx.x;   // b*L + pos
  if (idx >= B_*L_) return;
  int bb = idx / L_; int pos = idx - bb*L_;
  const float* xb = x + (size_t)bb*batchStride + pos;
  float mean = 0.f;
  for (int c = 0; c < C; ++c) mean += xb[(size_t)c*chanStride];
  mean /= C;
  float var = 0.f;
  for (int c = 0; c < C; ++c){ float d = xb[(size_t)c*chanStride]-mean; var += d*d; }
  var /= C;
  float inv = rsqrtf(var + 1e-5f);
  float* o = out + (size_t)idx*C;
  for (int c = 0; c < C; ++c)
    o[c] = (xb[(size_t)c*chanStride]-mean)*inv*g[c] + b[c];
}

// ------- dwconv 3x3 on xi = first DI cols of xz rows (NHWC) + SiLU -> NCHW -
__global__ void k_dwconv3x3_rows_silu(const float* __restrict__ xz,
                                      const float* __restrict__ w,
                                      const float* __restrict__ bias,
                                      float* __restrict__ out)
{
  int idx = blockIdx.x*blockDim.x + threadIdx.x;
  if (idx >= B_*DI*L_) return;
  int pos = idx % L_; int c = (idx/L_)%DI; int bb = idx/(DI*L_);
  int h = pos / W_, wc = pos % W_;
  float acc = bias[c];
  #pragma unroll
  for (int dy = -1; dy <= 1; ++dy){
    int hh = h+dy; if (hh < 0 || hh >= H_) continue;
    #pragma unroll
    for (int dx = -1; dx <= 1; ++dx){
      int w2 = wc+dx; if (w2 < 0 || w2 >= W_) continue;
      acc += w[c*9 + (dy+1)*3 + (dx+1)]
           * xz[((size_t)(bb*L_ + hh*W_ + w2))*(2*DI) + c];
    }
  }
  out[idx] = siluf_(acc);
}

// ---------------- build 4-direction xs from xconv (B,DI,L) ----------------
__global__ void k_build_xs(const float* __restrict__ xc, float* __restrict__ xs){
  int idx = blockIdx.x*blockDim.x + threadIdx.x;
  if (idx >= B_*DI*L_) return;
  int pos = idx % L_; int c = (idx/L_)%DI; int bb = idx/(DI*L_);
  float v = xc[idx];
  int h = pos / W_, w = pos % W_;
  int pt = w*H_ + h;
  size_t base = (size_t)bb*KDIR*DI*L_ + (size_t)c*L_;
  size_t ds = (size_t)DI*L_;
  xs[base + 0*ds + pos]          = v;
  xs[base + 1*ds + pt]           = v;
  xs[base + 2*ds + (L_-1-pos)]   = v;
  xs[base + 3*ds + (L_-1-pt)]    = v;
}

// ---------------- delta = softplus(dt_proj @ x_dbl[:R] + dt_bias) ----------
__global__ void k_dt_softplus(const float* __restrict__ xdbl,
                              const float* __restrict__ dtw,
                              const float* __restrict__ dtb,
                              float* __restrict__ delta)
{
  int idx = blockIdx.x*blockDim.x + threadIdx.x;
  if (idx >= B_*KDIR*DI*L_) return;
  int l  = idx % L_;
  int dc = (idx/L_) % DI;
  int k  = (idx/(DI*L_)) % KDIR;
  int bb = idx/(KDIR*DI*L_);
  const float* xr = xdbl + ((size_t)(bb*KDIR+k)*34)*L_ + l;
  float s = dtb[k*DI+dc];
  #pragma unroll
  for (int r = 0; r < RLOW; ++r)
    s += dtw[(k*DI+dc)*RLOW + r] * xr[(size_t)r*L_];
  delta[idx] = (s > 20.f) ? s : __logf(1.f + __expf(s));
}

__global__ void k_neg_exp(const float* __restrict__ a, float* __restrict__ o, int n){
  int i = blockIdx.x*blockDim.x + threadIdx.x;
  if (i < n) o[i] = -__expf(a[i]);
}

// ---------------- selective scan: block=(DI), grid=(B*KDIR) ----------------
// h[NST] lives in VGPRs; Bs/Cs columns broadcast through LDS each step.
__global__ void k_selective_scan(const float* __restrict__ u,
                                 const float* __restrict__ delta,
                                 const float* __restrict__ An,
                                 const float* __restrict__ xdbl,
                                 const float* __restrict__ Dp,
                                 float* __restrict__ y)
{
  int bk = blockIdx.x; int k = bk % KDIR;
  int dc = threadIdx.x;
  __shared__ float sB[NST], sC[NST];
  float h[NST], Ar[NST];
  #pragma unroll
  for (int n = 0; n < NST; ++n){ h[n] = 0.f; Ar[n] = An[((size_t)k*DI+dc)*NST+n]; }
  const float Dv = Dp[k*DI+dc];
  size_t chanBase = ((size_t)bk*DI + dc)*L_;
  const float* up = u + chanBase;
  const float* dp = delta + chanBase;
  float*       yp = y + chanBase;
  const float* Bp = xdbl + ((size_t)bk*34 + RLOW)*L_;
  const float* Cp = Bp + (size_t)NST*L_;
  for (int l = 0; l < L_; ++l){
    if (threadIdx.x < NST)        sB[threadIdx.x]      = Bp[(size_t)threadIdx.x*L_ + l];
    else if (threadIdx.x < 2*NST) sC[threadIdx.x-NST]  = Cp[(size_t)(threadIdx.x-NST)*L_ + l];
    __syncthreads();
    float ul = up[l], dl = dp[l];
    float du = dl*ul, acc = 0.f;
    #pragma unroll
    for (int n = 0; n < NST; ++n){
      h[n] = h[n]*__expf(dl*Ar[n]) + du*sB[n];
      acc += h[n]*sC[n];
    }
    yp[l] = acc + Dv*ul;
    __syncthreads();
  }
}

// ---------------- combine 4 scan directions -> rows (B*L, DI) --------------
__global__ void k_combine_y(const float* __restrict__ oy, float* __restrict__ yc){
  int idx = blockIdx.x*blockDim.x + threadIdx.x;
  if (idx >= B_*DI*L_) return;
  int pos = idx % L_; int c = (idx/L_)%DI; int bb = idx/(DI*L_);
  int h = pos / W_, w = pos % W_;
  int pt = w*H_ + h;
  size_t base = (size_t)bb*KDIR*DI*L_ + (size_t)c*L_;
  size_t ds = (size_t)DI*L_;
  float v = oy[base + pos]
          + oy[base + 2*ds + (L_-1-pos)]
          + oy[base + 1*ds + pt]
          + oy[base + 3*ds + (L_-1-pt)];
  yc[((size_t)(bb*L_+pos))*DI + c] = v;
}

// ---------------- out-LN over DI then gate by silu(z) (in place) -----------
__global__ void k_ln_gate(float* __restrict__ yc, const float* __restrict__ xz,
                          const float* __restrict__ g, const float* __restrict__ b)
{
  int idx = blockIdx.x*blockDim.x + threadIdx.x;
  if (idx >= B_*L_) return;
  float* row = yc + (size_t)idx*DI;
  const float* zr = xz + (size_t)idx*(2*DI) + DI;
  float mean = 0.f; for (int c = 0; c < DI; ++c) mean += row[c]; mean /= DI;
  float var = 0.f;  for (int c = 0; c < DI; ++c){ float d = row[c]-mean; var += d*d; } var /= DI;
  float inv = rsqrtf(var + 1e-5f);
  for (int c = 0; c < DI; ++c){
    float z = zr[c];
    row[c] = ((row[c]-mean)*inv*g[c] + b[c]) * (z*sigmoidf_(z));
  }
}

// ---------------- residual add: NCHW(out) = resid + rows(B*L,C) ------------
__global__ void k_add_rows_to_nchw(const float* __restrict__ rows,
                                   const float* __restrict__ resid,
                                   float* __restrict__ out, int C)
{
  int idx = blockIdx.x*blockDim.x + threadIdx.x;
  if (idx >= B_*C*L_) return;
  int pos = idx % L_; int c = (idx/L_)%C; int bb = idx/(C*L_);
  out[idx] = resid[idx] + rows[((size_t)(bb*L_+pos))*C + c];
}

// ---------------- PSA attention (cross-batch softmax per position) ---------
__global__ void k_psa_att(const float* __restrict__ x, const float* __restrict__ xh,
                          const float* __restrict__ xw, float* __restrict__ att)
{
  int pos = blockIdx.x*blockDim.x + threadIdx.x;
  if (pos >= L_) return;
  float s[B_], sh[B_], sw[B_];
  for (int bb = 0; bb < B_; ++bb){
    float a=0.f, b2=0.f, c2=0.f;
    size_t base = (size_t)bb*DBR*L_ + pos;
    for (int c = 0; c < DBR; ++c){
      a  += x [base + (size_t)c*L_];
      b2 += xh[base + (size_t)c*L_];
      c2 += xw[base + (size_t)c*L_];
    }
    s[bb]=a/DBR; sh[bb]=b2/DBR; sw[bb]=c2/DBR;
  }
  float eh[B_], ew[B_], mh=-1e30f, mw=-1e30f;
  for (int bb = 0; bb < B_; ++bb){
    eh[bb]=sh[bb]-s[bb]; ew[bb]=sw[bb]-s[bb];
    mh=fmaxf(mh,eh[bb]); mw=fmaxf(mw,ew[bb]);
  }
  float se=0.f, sw2=0.f;
  for (int bb = 0; bb < B_; ++bb){
    eh[bb]=__expf(eh[bb]-mh); ew[bb]=__expf(ew[bb]-mw); se+=eh[bb]; sw2+=ew[bb];
  }
  for (int bb = 0; bb < B_; ++bb){
    float rh = eh[bb]/se, rw = ew[bb]/sw2;
    // softmax(s-s)=1/B exactly -> s*(1+1/4)
    att[(size_t)bb*L_ + pos] =
      sigmoidf_(s[bb]*1.25f + sh[bb]*(1.f+rh) + sw[bb]*(1.f+rw));
  }
}

__global__ void k_mul_att_spatial(const float* __restrict__ x, const float* __restrict__ att,
                                  float* __restrict__ y){
  int idx = blockIdx.x*blockDim.x + threadIdx.x;
  if (idx >= B_*DBR*L_) return;
  int pos = idx % L_; int bb = idx/(DBR*L_);
  y[idx] = x[idx]*att[(size_t)bb*L_+pos];
}

// ---------------- generic NCHW depthwise conv (SAME) -----------------------
__global__ void k_dwconv_nchw(const float* __restrict__ x, const float* __restrict__ w,
                              const float* __restrict__ bias, float* __restrict__ y,
                              int C, int kh, int kw)
{
  int idx = blockIdx.x*blockDim.x + threadIdx.x;
  if (idx >= B_*C*L_) return;
  int pos = idx % L_; int c = (idx/L_)%C; int bb = idx/(C*L_);
  int h = pos/W_, ww = pos%W_;
  int ph = kh/2, pw = kw/2;
  float acc = bias ? bias[c] : 0.f;
  const float* wc = w + (size_t)c*kh*kw;
  const float* xb = x + (size_t)(bb*C + c)*L_;
  for (int i = 0; i < kh; ++i){
    int hh = h+i-ph; if (hh < 0 || hh >= H_) continue;
    for (int j = 0; j < kw; ++j){
      int w2 = ww+j-pw; if (w2 < 0 || w2 >= W_) continue;
      acc += wc[i*kw+j]*xb[hh*W_+w2];
    }
  }
  y[idx] = acc;
}

// ---------------- per-(b,c) spatial mean -----------------------------------
__global__ void k_mean_hw(const float* __restrict__ x, float* __restrict__ m){
  __shared__ float sm[256];
  int bc = blockIdx.x;
  const float* xb = x + (size_t)bc*L_;
  float a = 0.f;
  for (int i = threadIdx.x; i < L_; i += blockDim.x) a += xb[i];
  sm[threadIdx.x] = a; __syncthreads();
  for (int s = blockDim.x/2; s > 0; s >>= 1){
    if (threadIdx.x < s) sm[threadIdx.x] += sm[threadIdx.x+s];
    __syncthreads();
  }
  if (threadIdx.x == 0) m[bc] = sm[0]/(float)L_;
}

__global__ void k_pca_att(const float* __restrict__ cm, const float* __restrict__ cdm,
                          float* __restrict__ att){
  int bb = threadIdx.x; if (bb >= B_) return;
  float e[DBR]; float mx = -1e30f;
  for (int c = 0; c < DBR; ++c){ e[c] = cdm[bb*DBR+c]-cm[bb*DBR+c]; mx = fmaxf(mx,e[c]); }
  float s = 0.f;
  for (int c = 0; c < DBR; ++c){ e[c] = __expf(e[c]-mx); s += e[c]; }
  for (int c = 0; c < DBR; ++c){
    float r = e[c]/s;
    att[bb*DBR+c] = sigmoidf_(cdm[bb*DBR+c]*(1.f+r));
  }
}

__global__ void k_mul_att_channel(const float* __restrict__ x, const float* __restrict__ att,
                                  float* __restrict__ y){
  int idx = blockIdx.x*blockDim.x + threadIdx.x;
  if (idx >= B_*DBR*L_) return;
  int bc = idx / L_;
  y[idx] = x[idx]*att[bc];
}

// ---------------- instance norm (+ variants) -------------------------------
// mode 0: out = relu( IN(t + res) )
// mode 1: out = leaky0.01( IN(t) ) + res*scale
__global__ void k_instnorm(const float* __restrict__ t, long long tBS,
                           const float* __restrict__ res, long long resBS,
                           float* __restrict__ out, long long outBS,
                           const float* __restrict__ g, const float* __restrict__ bta,
                           int C, int mode, const float* __restrict__ scale)
{
  __shared__ float s1[256], s2[256];
  int bb = blockIdx.x / C, c = blockIdx.x % C;
  const float* tp = t + (size_t)bb*tBS + (size_t)c*L_;
  const float* rp = res ? (res + (size_t)bb*resBS + (size_t)c*L_) : nullptr;
  float a = 0.f, b2 = 0.f;
  for (int i = threadIdx.x; i < L_; i += blockDim.x){
    float v = tp[i]; if (mode == 0 && rp) v += rp[i];
    a += v; b2 += v*v;
  }
  s1[threadIdx.x] = a; s2[threadIdx.x] = b2; __syncthreads();
  for (int s = blockDim.x/2; s > 0; s >>= 1){
    if (threadIdx.x < s){ s1[threadIdx.x]+=s1[threadIdx.x+s]; s2[threadIdx.x]+=s2[threadIdx.x+s]; }
    __syncthreads();
  }
  float mean = s1[0]/(float)L_;
  float var  = s2[0]/(float)L_ - mean*mean;
  float inv  = rsqrtf(var + 1e-5f);
  float gg = g[c], bb2 = bta[c];
  float sc = scale ? scale[0] : 0.f;
  float* op = out + (size_t)bb*outBS + (size_t)c*L_;
  for (int i = threadIdx.x; i < L_; i += blockDim.x){
    float v = tp[i]; if (mode == 0 && rp) v += rp[i];
    float nv = (v-mean)*inv*gg + bb2;
    if (mode == 0) nv = fmaxf(nv, 0.f);
    else { nv = (nv >= 0.f) ? nv : 0.01f*nv; nv += rp[i]*sc; }
    op[i] = nv;
  }
}

// ---------------- axial: x + conv3x1 + conv1x3 (with biases) ---------------
__global__ void k_axial(const float* __restrict__ x, const float* __restrict__ wh,
                        const float* __restrict__ bh, const float* __restrict__ ww,
                        const float* __restrict__ bw, float* __restrict__ y, int C)
{
  int idx = blockIdx.x*blockDim.x + threadIdx.x;
  if (idx >= B_*C*L_) return;
  int pos = idx % L_; int c = (idx/L_)%C;
  int h = pos/W_, w = pos%W_;
  const float* xb = x + (idx - pos);
  float acc = x[idx] + bh[c] + bw[c];
  #pragma unroll
  for (int i = -1; i <= 1; ++i){ int hh=h+i; if (hh<0||hh>=H_) continue; acc += wh[c*3+(i+1)]*xb[hh*W_+w]; }
  #pragma unroll
  for (int j = -1; j <= 1; ++j){ int w2=w+j; if (w2<0||w2>=W_) continue; acc += ww[c*3+(j+1)]*xb[h*W_+w2]; }
  y[idx] = acc;
}

// ---------------- batch norm stats (per channel over B,HW) -----------------
__global__ void k_bn_stats(const float* __restrict__ x, float* __restrict__ mv, int C){
  __shared__ float s1[256], s2[256];
  int c = blockIdx.x;
  float a = 0.f, b2 = 0.f;
  for (int i = threadIdx.x; i < B_*L_; i += blockDim.x){
    int bb = i/L_, pos = i%L_;
    float v = x[((size_t)bb*C + c)*L_ + pos];
    a += v; b2 += v*v;
  }
  s1[threadIdx.x]=a; s2[threadIdx.x]=b2; __syncthreads();
  for (int s = blockDim.x/2; s > 0; s >>= 1){
    if (threadIdx.x < s){ s1[threadIdx.x]+=s1[threadIdx.x+s]; s2[threadIdx.x]+=s2[threadIdx.x+s]; }
    __syncthreads();
  }
  if (threadIdx.x == 0){
    float m = s1[0]/(float)(B_*L_);
    mv[2*c] = m; mv[2*c+1] = s2[0]/(float)(B_*L_) - m*m;
  }
}

__global__ void k_bn_apply_relu(float* __restrict__ x, const float* __restrict__ mv,
                                const float* __restrict__ g, const float* __restrict__ b, int C){
  int idx = blockIdx.x*blockDim.x + threadIdx.x;
  if (idx >= B_*C*L_) return;
  int c = (idx/L_)%C;
  float nv = (x[idx]-mv[2*c])*rsqrtf(mv[2*c+1]+1e-5f)*g[c]+b[c];
  x[idx] = fmaxf(nv, 0.f);
}

// ---------------- skip = bn3(xa) + bn2(xr) ---------------------------------
__global__ void k_skip(const float* __restrict__ xa, const float* __restrict__ mva,
                       const float* __restrict__ ga, const float* __restrict__ ba,
                       const float* __restrict__ xr, const float* __restrict__ mvr,
                       const float* __restrict__ gr, const float* __restrict__ br,
                       float* __restrict__ skip)
{
  int idx = blockIdx.x*blockDim.x + threadIdx.x;
  if (idx >= B_*COUT*L_) return;
  int c = (idx/L_)%COUT;
  float va = (xa[idx]-mva[2*c])*rsqrtf(mva[2*c+1]+1e-5f)*ga[c]+ba[c];
  float vr = (xr[idx]-mvr[2*c])*rsqrtf(mvr[2*c+1]+1e-5f)*gr[c]+br[c];
  skip[idx] = va + vr;
}

__global__ void k_maxpool_relu(const float* __restrict__ skip, float* __restrict__ out){
  int idx = blockIdx.x*blockDim.x + threadIdx.x;
  const int HW2 = (H_/2)*(W_/2);
  if (idx >= B_*COUT*HW2) return;
  int w = idx % (W_/2); int h = (idx/(W_/2)) % (H_/2); int bc = idx/HW2;
  const float* sp = skip + (size_t)bc*L_;
  float m = fmaxf(fmaxf(sp[(2*h)*W_+2*w],   sp[(2*h)*W_+2*w+1]),
                  fmaxf(sp[(2*h+1)*W_+2*w], sp[(2*h+1)*W_+2*w+1]));
  out[idx] = fmaxf(m, 0.f);
}

// ======================================================================
// Host orchestration
// ======================================================================
struct VssP {
  const float *A_log, *D, *conv_b, *conv_w, *dt_bias, *dt_proj_w, *in_proj;
  const float *ln_b, *ln_g, *out_ln_b, *out_ln_g, *out_proj, *x_proj_w;
};

struct VssWs {
  float *xn, *xz, *xconv, *xs, *xdbl, *delta, *aneg, *outy, *yc, *proj;
};

static inline dim3 g1(size_t n){ return dim3((unsigned)((n + 255)/256)); }

static void gemm(hipStream_t s, const float* A, const float* Bm, float* C,
                 int M, int N, int K, long long sA, long long sB, long long sC,
                 int batch, const float* bias, int biasMode, int transB)
{
  dim3 grid((N+TN-1)/TN, (M+TM-1)/TM, batch);
  k_gemm_wmma<<<grid, dim3(256), 0, s>>>(A, Bm, C, M, N, K, sA, sB, sC, bias, biasMode, transB);
}

static void run_vss(hipStream_t s, const VssP& p, const VssWs& w,
                    const float* in_nchw, float* out_nchw)
{
  // LN over channels -> rows (B*L, 32)
  k_ln_nchw_rows<<<g1(B_*L_),256,0,s>>>(in_nchw, L_, (long long)DBR*L_, p.ln_g, p.ln_b, w.xn, DBR);
  // xz = xn @ in_proj : (9216,32)x(32,128)
  gemm(s, w.xn, p.in_proj, w.xz, B_*L_, 2*DI, DBR, 0, 0, 0, 1, nullptr, 0, 0);
  // xconv = silu(dwconv3x3(xi)) in NCHW
  k_dwconv3x3_rows_silu<<<g1((size_t)B_*DI*L_),256,0,s>>>(w.xz, p.conv_w, p.conv_b, w.xconv);
  // 4 scan orders
  k_build_xs<<<g1((size_t)B_*DI*L_),256,0,s>>>(w.xconv, w.xs);
  // x_dbl per direction k: (34,64) x (64, L), batched over b
  for (int k = 0; k < KDIR; ++k){
    gemm(s, p.x_proj_w + (size_t)k*34*DI, w.xs + (size_t)k*DI*L_, w.xdbl + (size_t)k*34*L_,
         34, L_, DI,
         0, (long long)KDIR*DI*L_, (long long)KDIR*34*L_,
         B_, nullptr, 0, 0);
  }
  k_dt_softplus<<<g1((size_t)B_*KDIR*DI*L_),256,0,s>>>(w.xdbl, p.dt_proj_w, p.dt_bias, w.delta);
  k_neg_exp<<<g1(KDIR*DI*NST),256,0,s>>>(p.A_log, w.aneg, KDIR*DI*NST);
  k_selective_scan<<<dim3(B_*KDIR), dim3(DI), 0, s>>>(w.xs, w.delta, w.aneg, w.xdbl, p.D, w.outy);
  k_combine_y<<<g1((size_t)B_*DI*L_),256,0,s>>>(w.outy, w.yc);
  k_ln_gate<<<g1(B_*L_),256,0,s>>>(w.yc, w.xz, p.out_ln_g, p.out_ln_b);
  // proj = yc @ out_proj : (9216,64)x(64,32)
  gemm(s, w.yc, p.out_proj, w.proj, B_*L_, DBR, DI, 0, 0, 0, 1, nullptr, 0, 0);
  k_add_rows_to_nchw<<<g1((size_t)B_*DBR*L_),256,0,s>>>(w.proj, in_nchw, out_nchw, DBR);
}

static VssP load_vss(const float* const* in, int base){
  VssP p;
  p.A_log     = in[base+0];  p.D        = in[base+1];  p.conv_b   = in[base+2];
  p.conv_w    = in[base+3];  p.dt_bias  = in[base+4];  p.dt_proj_w= in[base+5];
  p.in_proj   = in[base+6];  p.ln_b     = in[base+7];  p.ln_g     = in[base+8];
  p.out_ln_b  = in[base+9];  p.out_ln_g = in[base+10]; p.out_proj = in[base+11];
  p.x_proj_w  = in[base+12];
  return p;
}

extern "C" void kernel_launch(void* const* d_in, const int* in_sizes, int n_in,
                              void* d_out_v, int out_size, void* d_ws_v, size_t ws_size,
                              hipStream_t stream)
{
  const float* const* in = (const float* const*)d_in;
  // Flattened pytree (sorted dict keys): params leaves [0..54], x last [55].
  const float* ax1_bh = in[0];  const float* ax1_bw = in[1];
  const float* ax1_wh = in[2];  const float* ax1_ww = in[3];
  const float* ax2_bh = in[4];  const float* ax2_bw = in[5];
  const float* ax2_wh = in[6];  const float* ax2_ww = in[7];
  const float* bn1_b  = in[8];  const float* bn1_g  = in[9];
  const float* bn2_b  = in[10]; const float* bn2_g  = in[11];
  const float* bn3_b  = in[12]; const float* bn3_g  = in[13];
  const float* conv1w = in[14]; const float* conv2w = in[15];
  const float* in1_b  = in[16]; const float* in1_g  = in[17];
  VssP mamba = load_vss(in, 18);
  const float* pca_b  = in[31]; const float* pca_w  = in[32];
  const float* psa_bh = in[33]; const float* psa_bw = in[34];
  const float* psa_wh = in[35]; const float* psa_ww = in[36];
  const float* rm_cb  = in[37]; const float* rm_cw  = in[38];
  const float* rm_inb = in[39]; const float* rm_ing = in[40];
  const float* rm_sc  = in[41];
  VssP rmvss = load_vss(in, 42);
  const float* x = in[n_in-1];   // 'x' sorts last

  float* out = (float*)d_out_v;
  float* skip = out + (size_t)B_*COUT*(H_/2)*(W_/2);   // tuple order: (out, skip)

  // ---------------- workspace carve-up ----------------
  float* ws = (float*)d_ws_v;
  size_t off = 0;
  auto take = [&](size_t n){ float* r = ws + off; off += n; return r; };
  float* WS_XC   = take((size_t)B_*CIN*L_);
  float* WS_T1   = take((size_t)B_*DBR*L_);
  float* WS_T2   = take((size_t)B_*DBR*L_);
  float* WS_T3   = take((size_t)B_*DBR*L_);
  float* WS_XH   = take((size_t)B_*DBR*L_);
  float* WS_XW   = take((size_t)B_*DBR*L_);
  float* WS_ATT  = take((size_t)B_*L_);
  float* WS_CM   = take((size_t)B_*DBR);
  float* WS_CDM  = take((size_t)B_*DBR);
  float* WS_CATT = take((size_t)B_*DBR);
  VssWs vw;
  vw.xn    = take((size_t)B_*L_*DBR);
  vw.xz    = take((size_t)B_*L_*2*DI);
  vw.xconv = take((size_t)B_*DI*L_);
  vw.xs    = take((size_t)B_*KDIR*DI*L_);
  vw.xdbl  = take((size_t)B_*KDIR*34*L_);
  vw.delta = take((size_t)B_*KDIR*DI*L_);
  vw.aneg  = take((size_t)KDIR*DI*NST);
  vw.outy  = take((size_t)B_*KDIR*DI*L_);
  vw.yc    = take((size_t)B_*L_*DI);
  vw.proj  = take((size_t)B_*L_*DBR);
  float* WS_XA  = take((size_t)B_*COUT*L_);
  float* WS_XR  = take((size_t)B_*COUT*L_);
  float* WS_MV1 = take(2*CIN);
  float* WS_MVA = take(2*COUT);
  float* WS_MVR = take(2*COUT);
  (void)ws_size; (void)in_sizes; (void)out_size;

  const long long BS32 = (long long)DBR*L_;   // batch stride of (B,32,L)
  const long long BSIN = (long long)CIN*L_;

  // ============ branches 0,1: psa -> pca -> vss(mamba) -> IN+relu ==========
  for (int br = 0; br < 2; ++br){
    int c0 = br*DBR;
    k_slice<<<g1((size_t)B_*DBR*L_),256,0,stream>>>(x, WS_T1, c0);
    // x_h: rows (b,c,h) x W  @ wh^T  (+bh per-N)
    gemm(stream, WS_T1, psa_wh, WS_XH, B_*DBR*H_, W_, W_, 0, 0, 0, 1, psa_bh, 1, 1);
    // x_w: per (b,c): ww (48x48) @ x[b,c] (48x48)  (+bw per-M)
    gemm(stream, psa_ww, WS_T1, WS_XW, H_, W_, H_, 0, (long long)L_, (long long)L_,
         B_*DBR, psa_bw, 2, 0);
    k_psa_att<<<g1(L_),256,0,stream>>>(WS_T1, WS_XH, WS_XW, WS_ATT);
    k_mul_att_spatial<<<g1((size_t)B_*DBR*L_),256,0,stream>>>(WS_T1, WS_ATT, WS_T2);
    // pca
    k_dwconv_nchw<<<g1((size_t)B_*DBR*L_),256,0,stream>>>(WS_T2, pca_w, pca_b, WS_T3, DBR, 9, 9);
    k_mean_hw<<<dim3(B_*DBR),dim3(256),0,stream>>>(WS_T2, WS_CM);
    k_mean_hw<<<dim3(B_*DBR),dim3(256),0,stream>>>(WS_T3, WS_CDM);
    k_pca_att<<<dim3(1),dim3(32),0,stream>>>(WS_CM, WS_CDM, WS_CATT);
    k_mul_att_channel<<<g1((size_t)B_*DBR*L_),256,0,stream>>>(WS_T3, WS_CATT, WS_T1);
    // vss block (residual on its own input)
    run_vss(stream, mamba, vw, WS_T1, WS_T2);
    // relu(IN(t + xb)) -> concat slot
    k_instnorm<<<dim3(B_*DBR),dim3(256),0,stream>>>(
        WS_T2, BS32, x + (size_t)c0*L_, BSIN,
        WS_XC + (size_t)c0*L_, BSIN, in1_g, in1_b, DBR, 0, nullptr);
  }

  // ============ branches 2,3: resmamba -> axial -> IN+relu =================
  for (int br = 2; br < 4; ++br){
    int c0 = br*DBR;
    const float* awh = (br==2)? ax1_wh : ax2_wh;
    const float* abh = (br==2)? ax1_bh : ax2_bh;
    const float* aww = (br==2)? ax1_ww : ax2_ww;
    const float* abw = (br==2)? ax1_bw : ax2_bw;
    k_slice<<<g1((size_t)B_*DBR*L_),256,0,stream>>>(x, WS_T1, c0);
    // resmamba
    k_dwconv_nchw<<<g1((size_t)B_*DBR*L_),256,0,stream>>>(WS_T1, rm_cw, rm_cb, WS_T2, DBR, 3, 3);
    run_vss(stream, rmvss, vw, WS_T2, WS_T3);
    k_instnorm<<<dim3(B_*DBR),dim3(256),0,stream>>>(
        WS_T3, BS32, WS_T1, BS32, WS_T2, BS32, rm_ing, rm_inb, DBR, 1, rm_sc);
    // axial
    k_axial<<<g1((size_t)B_*DBR*L_),256,0,stream>>>(WS_T2, awh, abh, aww, abw, WS_T3, DBR);
    // relu(IN(t + xb)) -> concat slot
    k_instnorm<<<dim3(B_*DBR),dim3(256),0,stream>>>(
        WS_T3, BS32, x + (size_t)c0*L_, BSIN,
        WS_XC + (size_t)c0*L_, BSIN, in1_g, in1_b, DBR, 0, nullptr);
  }

  // ============ head: bn1+relu, 1x1 convs (WMMA), bn2/bn3, pool ============
  k_bn_stats<<<dim3(CIN),dim3(256),0,stream>>>(WS_XC, WS_MV1, CIN);
  k_bn_apply_relu<<<g1((size_t)B_*CIN*L_),256,0,stream>>>(WS_XC, WS_MV1, bn1_g, bn1_b, CIN);
  // xa = conv1_w (256x128) @ xc[b] (128xL), batched over b
  gemm(stream, conv1w, WS_XC, WS_XA, COUT, L_, CIN, 0, BSIN, (long long)COUT*L_, B_, nullptr, 0, 0);
  // xr = conv2_w @ x[b]
  gemm(stream, conv2w, x, WS_XR, COUT, L_, CIN, 0, BSIN, (long long)COUT*L_, B_, nullptr, 0, 0);
  k_bn_stats<<<dim3(COUT),dim3(256),0,stream>>>(WS_XA, WS_MVA, COUT);
  k_bn_stats<<<dim3(COUT),dim3(256),0,stream>>>(WS_XR, WS_MVR, COUT);
  k_skip<<<g1((size_t)B_*COUT*L_),256,0,stream>>>(WS_XA, WS_MVA, bn3_g, bn3_b,
                                                  WS_XR, WS_MVR, bn2_g, bn2_b, skip);
  k_maxpool_relu<<<g1((size_t)B_*COUT*(H_/2)*(W_/2)),256,0,stream>>>(skip, out);
}
